// SelfAttention_27685359190700
// MI455X (gfx1250) — matrix operand order repaired and checked by reading, not compiled, not run on hardware
//
#include <hip/hip_runtime.h>
#include <hip/hip_bf16.h>

typedef __attribute__((ext_vector_type(16))) _Float16 v16h;
typedef __attribute__((ext_vector_type(8)))  _Float16 v8h;
typedef __attribute__((ext_vector_type(8)))  float    v8f;

#define BATCH 8
#define CCH   256
#define DCH   32
#define NTOK  4096
#define SPAD  4100   // padded LDS row stride (floats) for the 16x4096 score tile
#define XPAD  40     // padded token-major LDS row stride (f16) for staged x tile

union V16U { v16h v; v8h h[2]; };

__device__ __forceinline__ v8f wmma_f16(v16h a, v16h b, v8f c) {
  // D = A(16x32 f16) * B(32x16 f16) + C(16x16 f32)
  return __builtin_amdgcn_wmma_f32_16x16x32_f16(
      /*neg_a=*/false, a, /*neg_b=*/false, b,
      /*c_mod=*/(short)0, c, /*reuse_a=*/false, /*reuse_b=*/false);
}

// A-matrix (16x32, 16-bit) element->K map per ISA 7.12.2:
// lanes 0-15 hold K=0..7 (elems 0..7) and K=16..23 (elems 8..15);
// lanes 16-31 hold K=8..15 and K=24..31.
__device__ __forceinline__ int kmapA(int e, int hi) {
  return e + 8 * hi + ((e >= 8) ? 8 : 0);
}

// ---------------------------------------------------------------------------
// Kernel 1: fused QKV projection.  Y[320,4096] = W[320,256] * X[256,4096] (+b)
// x tile staged through LDS token-major (transpose on the write side), so the
// WMMA B operand is two 16B vector LDS loads per lane instead of 16 scalars.
// Q,K stored token-major f16 [b][n][32]; V channel-major f16 [b][c][n].
// ---------------------------------------------------------------------------
__global__ void qkv_proj_kernel(const float* __restrict__ x,
                                const float* __restrict__ wq, const float* __restrict__ bq,
                                const float* __restrict__ wk, const float* __restrict__ bk,
                                const float* __restrict__ wv, const float* __restrict__ bv,
                                _Float16* __restrict__ Qh, _Float16* __restrict__ Kh,
                                _Float16* __restrict__ Vh) {
  __shared__ _Float16 Xs[64 * XPAD];        // [token 0..63][k 0..31], padded

  const int wave = threadIdx.x >> 5;        // wave32
  const int lane = threadIdx.x & 31;
  const int lo = lane & 15, hi = lane >> 4;
  const int nb0 = blockIdx.x * 64;          // 4 waves x 16 tokens
  const int m0 = blockIdx.y * 16;
  const int bi = blockIdx.z;
  const int n  = nb0 + wave * 16 + lo;
  const float* xb = x + (size_t)bi * CCH * NTOK;

  // per-lane A row (stacked weight matrix row; weights are L2-resident)
  const int row = m0 + lo;
  const float* wrow;
  if (row < 32)      wrow = wq + row * CCH;
  else if (row < 64) wrow = wk + (row - 32) * CCH;
  else               wrow = wv + (size_t)(row - 64) * CCH;

  v8f acc = {};
  for (int k0 = 0; k0 < CCH; k0 += 32) {
    // stage 32(k) x 64(n): global reads coalesced along n; LDS writes do the
    // k<->n transpose (scalar b16, ~2-way bank conflicts with 80B stride)
#pragma unroll
    for (int p = 0; p < 16; ++p) {
      const int idx = threadIdx.x + 128 * p;
      const int kk = idx >> 6, nn = idx & 63;
      Xs[nn * XPAD + kk] = (_Float16)xb[(size_t)(k0 + kk) * NTOK + nb0 + nn];
    }
    __syncthreads();
    v16h a;
#pragma unroll
    for (int e = 0; e < 16; ++e)
      a[e] = (_Float16)wrow[k0 + 16 * hi + e];
    // B-matrix (32x16): lane = column n, K = e + 16*hi -> contiguous in Xs row
    V16U b;
    const _Float16* xs = Xs + (wave * 16 + lo) * XPAD + 16 * hi;
    b.h[0] = *(const v8h*)(xs);              // 16B LDS vector load
    b.h[1] = *(const v8h*)(xs + 8);          // 16B LDS vector load
    acc = wmma_f16(a, b.v, acc);
    __syncthreads();
  }

  // C layout: VGPR g, lanes<16 -> row m0+g ; lanes>=16 -> row m0+8+g
  if (m0 < 64) {
    const bool   isQ = (m0 < 32);
    _Float16*    dst = isQ ? Qh : Kh;
    const float* bia = isQ ? bq : bk;
    const int    mb  = isQ ? m0 : (m0 - 32);
    v8h o;
#pragma unroll
    for (int g = 0; g < 8; ++g)
      o[g] = (_Float16)(acc[g] + bia[mb + 8 * hi + g]);
    *(v8h*)(dst + ((size_t)(bi * NTOK + n)) * DCH + mb + 8 * hi) = o;   // 16B store
  } else {
    const int cb = m0 - 64;
#pragma unroll
    for (int g = 0; g < 8; ++g) {
      const int c = cb + 8 * hi + g;
      Vh[((size_t)(bi * CCH + c)) * NTOK + n] = (_Float16)(acc[g] + bv[c]);
    }
  }
}

// ---------------------------------------------------------------------------
// Kernel 2: scores (K^T Q, inner dim d=32 = one WMMA k-step) + row softmax.
// Workgroup = 16 rows x 4096 cols staged in ~258 KB LDS (fits 320 KB WGP LDS).
// Output phase: lane-adjacent = address-adjacent (coalesced float4 stores of
// the f32 attention map) + register-packed 32B v16h stores of A^T (f16).
// ---------------------------------------------------------------------------
__global__ void attn_softmax_kernel(const _Float16* __restrict__ Qh,
                                    const _Float16* __restrict__ Kh,
                                    float* __restrict__ Amap,
                                    _Float16* __restrict__ AT) {
  extern __shared__ float lds[];
  float* S    = lds;                 // 16 x SPAD scores
  float* red  = lds + 16 * SPAD;     // 16 x 16 partials
  float* rowv = red + 256;           // 16 row stats

  const int wave = threadIdx.x >> 5;
  const int lane = threadIdx.x & 31;
  const int lo = lane & 15, hi = lane >> 4;
  const int i0 = blockIdx.x * 16;
  const int bi = blockIdx.y;

  // A operand: K^T tile = 16 tokens (rows) x 32 d (token-major f16)
  const _Float16* kp = Kh + ((size_t)(bi * NTOK + i0 + lo)) * DCH;
  V16U a;
  a.v = *(const v16h*)(kp + 16 * hi);

  for (int t = 0; t < 32; ++t) {                 // 8 waves x 32 tiles = 4096 cols
    const int j0 = wave * 512 + t * 16;
    const _Float16* qp = Qh + ((size_t)(bi * NTOK + j0 + lo)) * DCH;
    v16h b = *(const v16h*)(qp + 16 * hi);       // 32B contiguous per lane
    v8f z = {};
    v8f c = wmma_f16(a.v, b, z);
#pragma unroll
    for (int g = 0; g < 8; ++g)
      S[(g + 8 * hi) * SPAD + j0 + lo] = c[g];
  }
  __syncthreads();

  // 16 threads cooperate per row; each owns a contiguous 256-col chunk (f4 ops)
  const int r   = threadIdx.x & 15;
  const int sub = threadIdx.x >> 4;
  float* Sr = S + r * SPAD + sub * 256;

  float m = -3.402823466e38f;
  for (int k = 0; k < 256; k += 4) {
    const float4 v = *(const float4*)(Sr + k);
    m = fmaxf(m, fmaxf(fmaxf(v.x, v.y), fmaxf(v.z, v.w)));
  }
  red[r * 16 + sub] = m;
  __syncthreads();
  if (sub == 0) {
    float mm = red[r * 16];
    for (int k = 1; k < 16; ++k) mm = fmaxf(mm, red[r * 16 + k]);
    rowv[r] = mm;
  }
  __syncthreads();
  const float rmax = rowv[r];
  float s = 0.0f;
  for (int k = 0; k < 256; k += 4) {
    float4 v = *(float4*)(Sr + k);
    v.x = __expf(v.x - rmax);
    v.y = __expf(v.y - rmax);
    v.z = __expf(v.z - rmax);
    v.w = __expf(v.w - rmax);
    s += (v.x + v.y) + (v.z + v.w);
    *(float4*)(Sr + k) = v;
  }
  red[r * 16 + sub] = s;
  __syncthreads();
  if (sub == 0) {
    float ss = 0.0f;
    for (int k = 0; k < 16; ++k) ss += red[r * 16 + k];
    rowv[r] = ss;
  }
  __syncthreads();

  // Output: thread owns 4 consecutive columns per pass -> coalesced stores.
  float invv[16];
#pragma unroll
  for (int rr = 0; rr < 16; ++rr) invv[rr] = 1.0f / rowv[rr];

  for (int jb = 0; jb < 4; ++jb) {
    const int j4 = jb * 1024 + threadIdx.x * 4;
    v16h p0, p1, p2, p3;                         // A^T pack: 16 rows per column
#pragma unroll
    for (int rr = 0; rr < 16; ++rr) {
      float4 v = *(const float4*)&S[rr * SPAD + j4];
      v.x *= invv[rr]; v.y *= invv[rr]; v.z *= invv[rr]; v.w *= invv[rr];
      *(float4*)&Amap[((size_t)(bi * NTOK + i0 + rr)) * NTOK + j4] = v;
      p0[rr] = (_Float16)v.x; p1[rr] = (_Float16)v.y;
      p2[rr] = (_Float16)v.z; p3[rr] = (_Float16)v.w;
    }
    _Float16* tb = AT + ((size_t)(bi * NTOK + j4)) * NTOK + i0;
    *(v16h*)(tb)             = p0;               // 32B contiguous per column
    *(v16h*)(tb + NTOK)      = p1;
    *(v16h*)(tb + 2 * NTOK)  = p2;
    *(v16h*)(tb + 3 * NTOK)  = p3;
  }
}

// ---------------------------------------------------------------------------
// Kernel 3: out^T[n,c] = sum_m A^T[n,m] * V^T[m,c]; epilogue gamma*att + x.
// Each wave: one n-tile x FOUR c-tiles sharing the A operand -> 4 independent
// WMMA accumulation chains per k-step, 4x A-traffic amortization.
// ---------------------------------------------------------------------------
__global__ void av_out_kernel(const _Float16* __restrict__ Vh,
                              const _Float16* __restrict__ AT,
                              const float* __restrict__ x,
                              const float* __restrict__ gamma,
                              float* __restrict__ out) {
  const int wave = threadIdx.x >> 5;
  const int lane = threadIdx.x & 31;
  const int lo = lane & 15, hi = lane >> 4;
  const int n0 = blockIdx.x * 16;
  const int bi = blockIdx.y;
  const int cbase = wave * 64;                   // 4 waves x 64 ch = 256 = C

  const _Float16* ap  = AT + ((size_t)(bi * NTOK + n0 + lo)) * NTOK;   // m-contig
  const _Float16* vp0 = Vh + ((size_t)(bi * CCH + cbase + lo)) * NTOK; // m-contig

  v8f acc0 = {}, acc1 = {}, acc2 = {}, acc3 = {};
  for (int m0 = 0; m0 < NTOK; m0 += 32) {
    V16U a;
    a.v = *(const v16h*)(ap + m0 + 16 * hi);
    __builtin_prefetch(ap + m0 + 512, 0, 1);     // global_prefetch_b8
    const v16h b0 = *(const v16h*)(vp0 + m0 + 16 * hi);
    const v16h b1 = *(const v16h*)(vp0 + (size_t)16 * NTOK + m0 + 16 * hi);
    const v16h b2 = *(const v16h*)(vp0 + (size_t)32 * NTOK + m0 + 16 * hi);
    const v16h b3 = *(const v16h*)(vp0 + (size_t)48 * NTOK + m0 + 16 * hi);
    __builtin_prefetch(vp0 + m0 + 512, 0, 1);
    acc0 = wmma_f16(a.v, b0, acc0);
    acc1 = wmma_f16(a.v, b1, acc1);
    acc2 = wmma_f16(a.v, b2, acc2);
    acc3 = wmma_f16(a.v, b3, acc3);
  }

  const float gm = gamma[0];
#pragma unroll
  for (int u = 0; u < 4; ++u) {
    const v8f acc = (u == 0) ? acc0 : (u == 1) ? acc1 : (u == 2) ? acc2 : acc3;
    const int c = cbase + u * 16 + lo;
    const size_t base = ((size_t)(bi * CCH + c)) * NTOK + n0 + 8 * hi;
    const float* xp = x + base;
    float*       op = out + base;
#pragma unroll
    for (int g = 0; g < 8; ++g)
      op[g] = gm * acc[g] + xp[g];               // 8 consecutive f32 per lane
  }
}

// ---------------------------------------------------------------------------
extern "C" void kernel_launch(void* const* d_in, const int* in_sizes, int n_in,
                              void* d_out, int out_size, void* d_ws, size_t ws_size,
                              hipStream_t stream) {
  const float* x     = (const float*)d_in[0];
  const float* wq    = (const float*)d_in[1];
  const float* bq    = (const float*)d_in[2];
  const float* wk    = (const float*)d_in[3];
  const float* bk    = (const float*)d_in[4];
  const float* wv    = (const float*)d_in[5];
  const float* bv    = (const float*)d_in[6];
  const float* gamma = (const float*)d_in[7];

  float* out  = (float*)d_out;                          // [B,C,W,H]
  float* Amap = out + (size_t)BATCH * CCH * NTOK;       // [B,N,N]

  char* ws = (char*)d_ws;
  _Float16* Qh = (_Float16*)(ws);                          //  2 MB
  _Float16* Kh = (_Float16*)(ws + (((size_t)2)  << 20));   //  2 MB
  _Float16* Vh = (_Float16*)(ws + (((size_t)4)  << 20));   // 16 MB
  _Float16* AT = (_Float16*)(ws + (((size_t)20) << 20));   // 268 MB

  dim3 g1(NTOK / 64, 320 / 16, BATCH);
  qkv_proj_kernel<<<g1, dim3(128), 0, stream>>>(x, wq, bq, wk, bk, wv, bv, Qh, Kh, Vh);

  const size_t sh2 = (size_t)(16 * SPAD + 256 + 16) * sizeof(float);  // ~258 KB LDS
  dim3 g2(NTOK / 16, BATCH);
  attn_softmax_kernel<<<g2, dim3(256), sh2, stream>>>(Qh, Kh, Amap, AT);

  dim3 g3(NTOK / 16, BATCH);
  av_out_kernel<<<g3, dim3(128), 0, stream>>>(Vh, AT, x, gamma, out);
}
